// Attention_7413113553606
// MI455X (gfx1250) — matrix operand order repaired and checked by reading, not compiled, not run on hardware
//
#include <hip/hip_runtime.h>

typedef __attribute__((ext_vector_type(16))) __bf16 v16bf;
typedef __attribute__((ext_vector_type(8)))  float  v8f;
typedef unsigned int v4u __attribute__((ext_vector_type(4)));
typedef int v8i_t __attribute__((ext_vector_type(8)));
typedef int v4i_t __attribute__((ext_vector_type(4)));

#define TSEQ   2048
#define DMODEL 1024
#define EDIM   1024
#define NBATCH 8

#if defined(__gfx1250__) && __has_builtin(__builtin_amdgcn_tensor_load_to_lds) && \
    __has_builtin(__builtin_amdgcn_s_wait_tensorcnt)
#define USE_TDM 1
#else
#define USE_TDM 0
#endif

// f32 -> bf16 via native conversion (backend picks v_cvt_*_bf16 path)
__device__ __forceinline__ unsigned short f2bf(float f) {
  union { __bf16 b; unsigned short s; } c;
  c.b = (__bf16)f;
  return c.s;
}

__device__ __forceinline__ v8f wmma_bf16(v16bf a, v16bf b, v8f c) {
  // (neg_a, A, neg_b, B, c_mod, C, reuse_a, reuse_b)
  return __builtin_amdgcn_wmma_f32_16x16x32_bf16(false, a, false, b, (short)0, c,
                                                 false, false);
}

#if USE_TDM
// Issue a TDM 2-D tile load: tile_h rows x tile_w bf16 elems, global row
// stride row_stride elems, into LDS at lds_off with 1-DWORD pad per 64 DWORDs
// (so a 128-elem row lands on a 65-uint stride, matching the fragment loaders).
__device__ __forceinline__ void tdm_load_2d(unsigned lds_off, const void* gptr,
                                            unsigned tile_w, unsigned tile_h,
                                            unsigned row_stride) {
  unsigned long long ga = (unsigned long long)gptr;
  v4u g0;
  g0.x = 1u;                                            // count=1 valid descriptor
  g0.y = lds_off;                                       // lds_addr (bytes)
  g0.z = (unsigned)ga;                                  // global_addr[31:0]
  g0.w = (unsigned)((ga >> 32) & 0x01FFFFFFull) | (2u << 30);  // addr[56:32]|type=2
  v8i_t g1;
  // data_size=1 (2B), pad_enable=1, pad_interval=5 (64 DW), pad_amount=0 (1 DW)
  g1[0] = (int)((1u << 16) | (1u << 20) | (5u << 22));
  g1[1] = (int)((row_stride & 0xFFFFu) << 16);          // tensor_dim0[15:0], bar_addr=0
  g1[2] = 0;                                            // dim0[31:16]=0, dim1[15:0]=0
  g1[3] = (int)(0x100u | (tile_w << 16));               // dim1=1<<24, tile_dim0
  g1[4] = (int)(tile_h & 0xFFFFu);                      // tile_dim1, tile_dim2=0
  g1[5] = (int)row_stride;                              // tensor_dim0_stride[31:0]
  g1[6] = 0;
  g1[7] = 0;
  v4i_t z4 = {};
#if __clang_major__ >= 23
  v8i_t z8 = {};
  __builtin_amdgcn_tensor_load_to_lds(g0, g1, z4, z4, z8, 0);
#else
  __builtin_amdgcn_tensor_load_to_lds(g0, g1, z4, z4, 0);
#endif
}
#define LDSOFF(p) ((unsigned)(unsigned long long)(const void*)(p))
#endif

// A fragment 16(M)x32(K) bf16 from LDS laid out [row][k] (uint = bf16 pair).
__device__ __forceinline__ v16bf frag_a(const unsigned int* lds, int row_base,
                                        int stride_u, int k_off_u, int lane) {
  int m = lane & 15, half = lane >> 4;
  union { v16bf v; unsigned int u[8]; } f;
  const unsigned int* p = lds + (row_base + m) * stride_u + k_off_u + half * 8;
  f.u[0] = p[0]; f.u[1] = p[1]; f.u[2] = p[2];  f.u[3] = p[3];
  f.u[4] = p[4]; f.u[5] = p[5]; f.u[6] = p[6]; f.u[7] = p[7];
  return f.v;
}

// B fragment 32(K)x16(N) bf16 from LDS laid out [n][k] (k contiguous).
__device__ __forceinline__ v16bf frag_b(const unsigned int* lds, int n_base,
                                        int stride_u, int k_off_u, int lane) {
  int n = lane & 15, half = lane >> 4;
  union { v16bf v; unsigned int u[8]; } f;
  const unsigned int* p = lds + (n_base + n) * stride_u + k_off_u + half * 8;
#pragma unroll
  for (int r = 0; r < 8; ++r) f.u[r] = p[r];
  return f.v;
}

// B fragment 32(K)x16(N) from LDS laid out [k][n] (k strided, stride in ushorts).
__device__ __forceinline__ v16bf frag_b_strided(const unsigned short* lds,
                                                int n_base, int stride_h, int lane) {
  int n = lane & 15, half = lane >> 4;
  union { v16bf v; unsigned short h[16]; } f;
  const unsigned short* p = lds + (half * 16) * stride_h + n_base + n;
#pragma unroll
  for (int e = 0; e < 16; ++e) f.h[e] = p[e * stride_h];
  return f.v;
}

// ---------------------------------------------------------------------------
// Kernel 1: Q/K/V projection.  C[16384,1024] = X[16384,1024] @ W[1024,1024]
// Tile 128x64, BK=32, 8 waves each computing a 32x32 sub-tile (2x2 WMMA).
// ---------------------------------------------------------------------------
__global__ __launch_bounds__(256) void qkv_proj_kernel(
    const float* __restrict__ x,  const float* __restrict__ Wq,
    const float* __restrict__ Wk, const float* __restrict__ Wv,
    unsigned short* __restrict__ Q, unsigned short* __restrict__ K,
    unsigned short* __restrict__ V) {
  __shared__ unsigned int lds_x[128 * 17];  // [row][k/2], ushort stride 34
  __shared__ unsigned int lds_w[64 * 17];   // [n][k/2]   (W transposed)
  const int t = threadIdx.x;
  const int lane = t & 31, w = t >> 5;
  const int m0 = blockIdx.x * 128;
  const int n0 = blockIdx.y * 64;
  const float* W = (blockIdx.z == 0) ? Wq : (blockIdx.z == 1) ? Wk : Wv;
  unsigned short* out = (blockIdx.z == 0) ? Q : (blockIdx.z == 1) ? K : V;
  const int wr = w & 3, wc = w >> 2;

  v8f z = {};
  v8f acc[2][2] = {{z, z}, {z, z}};

  for (int k0 = 0; k0 < DMODEL; k0 += 32) {
    unsigned short* xh = (unsigned short*)lds_x;
    unsigned short* wh = (unsigned short*)lds_w;
#pragma unroll
    for (int i = 0; i < 16; ++i) {            // 128x32 X tile
      int li = t + 256 * i;
      int row = li >> 5, col = li & 31;
      xh[row * 34 + col] = f2bf(x[(size_t)(m0 + row) * DMODEL + k0 + col]);
    }
#pragma unroll
    for (int i = 0; i < 8; ++i) {             // 32x64 W tile -> transposed
      int li = t + 256 * i;
      int kk = li >> 6, nn = li & 63;
      wh[nn * 34 + kk] = f2bf(W[(size_t)(k0 + kk) * EDIM + n0 + nn]);
    }
    __syncthreads();
    v16bf a0 = frag_a(lds_x, wr * 32,      17, 0, lane);
    v16bf a1 = frag_a(lds_x, wr * 32 + 16, 17, 0, lane);
    v16bf b0 = frag_b(lds_w, wc * 32,      17, 0, lane);
    v16bf b1 = frag_b(lds_w, wc * 32 + 16, 17, 0, lane);
    acc[0][0] = wmma_bf16(a0, b0, acc[0][0]);
    acc[0][1] = wmma_bf16(a0, b1, acc[0][1]);
    acc[1][0] = wmma_bf16(a1, b0, acc[1][0]);
    acc[1][1] = wmma_bf16(a1, b1, acc[1][1]);
    __syncthreads();
  }

  const int nl = lane & 15, half = lane >> 4;
#pragma unroll
  for (int i = 0; i < 2; ++i)
#pragma unroll
    for (int j = 0; j < 2; ++j)
#pragma unroll
      for (int r = 0; r < 8; ++r) {
        int row = m0 + wr * 32 + i * 16 + r + half * 8;
        int col = n0 + wc * 32 + j * 16 + nl;
        out[(size_t)row * EDIM + col] = f2bf(acc[i][j][r]);
      }
}

// ---------------------------------------------------------------------------
// Kernel 2: per-row softmax statistics (m = rowmax, l = sum exp), causal.
// ---------------------------------------------------------------------------
__global__ __launch_bounds__(256) void attn_stats_kernel(
    const unsigned short* __restrict__ Q, const unsigned short* __restrict__ K,
    float* __restrict__ Mrow, float* __restrict__ Lrow) {
  __shared__ unsigned int lds_q[64 * 65];   // [row][e/2], E chunk of 128 (TDM pad)
  __shared__ unsigned int lds_k[32 * 65];
  __shared__ float        lds_s[64 * 33];
  const int t = threadIdx.x;
  const int lane = t & 31, w = t >> 5;
  const int r0 = blockIdx.x * 64;
  const int b  = blockIdx.y;
  const int sm = w & 3, sn = w >> 2;
  const size_t qrow0 = (size_t)b * TSEQ + r0;

  float m_run = -3.0e38f, l_run = 0.f;

  for (int j = 0; j < r0 + 64; j += 32) {
    v8f s = {};
    for (int ec = 0; ec < EDIM; ec += 128) {
#if USE_TDM
      if (w == 0) {
        tdm_load_2d(LDSOFF(lds_q), Q + qrow0 * EDIM + ec, 128, 64, EDIM);
        tdm_load_2d(LDSOFF(lds_k), K + ((size_t)b * TSEQ + j) * EDIM + ec,
                    128, 32, EDIM);
        __builtin_amdgcn_s_wait_tensorcnt(0);
      }
#else
      {
        const unsigned int* Qu = (const unsigned int*)Q;
        const unsigned int* Ku = (const unsigned int*)K;
#pragma unroll
        for (int i = 0; i < 16; ++i) {
          int li = t + 256 * i;
          int row = li >> 6, cu = li & 63;
          lds_q[row * 65 + cu] =
              Qu[(qrow0 + row) * (EDIM / 2) + ec / 2 + cu];
        }
#pragma unroll
        for (int i = 0; i < 8; ++i) {
          int li = t + 256 * i;
          int key = li >> 6, cu = li & 63;
          lds_k[key * 65 + cu] =
              Ku[((size_t)b * TSEQ + j + key) * (EDIM / 2) + ec / 2 + cu];
        }
      }
#endif
      __syncthreads();
#pragma unroll
      for (int kk = 0; kk < 4; ++kk) {
        v16bf a  = frag_a(lds_q, sm * 16, 65, kk * 16, lane);
        v16bf bb = frag_b(lds_k, sn * 16, 65, kk * 16, lane);
        s = wmma_bf16(a, bb, s);
      }
      __syncthreads();
    }
    const int nl = lane & 15, half = lane >> 4;
#pragma unroll
    for (int r = 0; r < 8; ++r) {
      int row = sm * 16 + r + half * 8;       // local row 0..63
      int key = j + sn * 16 + nl;
      float sv = s[r] * 0.03125f;             // 1/sqrt(1024)
      if (key > r0 + row) sv = -3.0e38f;      // causal mask
      lds_s[row * 33 + sn * 16 + nl] = sv;
    }
    __syncthreads();
    if (t < 64) {                              // online max/sum per row
      float mj = -3.0e38f;
#pragma unroll
      for (int c = 0; c < 32; ++c) mj = fmaxf(mj, lds_s[t * 33 + c]);
      float mnew = fmaxf(m_run, mj);
      float sum = 0.f;
#pragma unroll
      for (int c = 0; c < 32; ++c) sum += __expf(lds_s[t * 33 + c] - mnew);
      l_run = l_run * __expf(m_run - mnew) + sum;
      m_run = mnew;
    }
    __syncthreads();
  }
  if (t < 64) {
    Mrow[(size_t)b * TSEQ + r0 + t] = m_run;
    Lrow[(size_t)b * TSEQ + r0 + t] = l_run;
  }
}

// ---------------------------------------------------------------------------
// Kernel 3: O = softmax(S) @ V, using precomputed (m, l).
// ---------------------------------------------------------------------------
__global__ __launch_bounds__(256) void attn_out_kernel(
    const unsigned short* __restrict__ Q, const unsigned short* __restrict__ K,
    const unsigned short* __restrict__ V, const float* __restrict__ Mrow,
    const float* __restrict__ Lrow, float* __restrict__ out) {
  __shared__ unsigned int lds_q[64 * 65];
  __shared__ unsigned int lds_k[32 * 65];
  __shared__ unsigned int lds_v[32 * 65];   // [key][col], ushort stride 130
  __shared__ unsigned int lds_p[64 * 17];   // [row][key], ushort stride 34
  __shared__ float lds_m[64];
  __shared__ float lds_li[64];
  const int t = threadIdx.x;
  const int lane = t & 31, w = t >> 5;
  const int c0 = blockIdx.x * 128;
  const int r0 = blockIdx.y * 64;
  const int b  = blockIdx.z;
  const int sm = w & 3, sn = w >> 2;
  const size_t qrow0 = (size_t)b * TSEQ + r0;

  if (t < 64) {
    lds_m[t]  = Mrow[(size_t)b * TSEQ + r0 + t];
    lds_li[t] = 1.0f / Lrow[(size_t)b * TSEQ + r0 + t];
  }
  __syncthreads();

  v8f z = {};
  v8f acc[4] = {z, z, z, z};

  for (int j = 0; j < r0 + 64; j += 32) {
#if USE_TDM
    if (w == 0)
      tdm_load_2d(LDSOFF(lds_v), V + ((size_t)b * TSEQ + j) * EDIM + c0,
                  128, 32, EDIM);
#else
    {
      const unsigned int* Vu = (const unsigned int*)V;
#pragma unroll
      for (int i = 0; i < 8; ++i) {
        int li = t + 256 * i;
        int key = li >> 6, cu = li & 63;
        lds_v[key * 65 + cu] =
            Vu[((size_t)b * TSEQ + j + key) * (EDIM / 2) + c0 / 2 + cu];
      }
    }
#endif
    v8f s = {};
    for (int ec = 0; ec < EDIM; ec += 128) {
#if USE_TDM
      if (w == 0) {
        tdm_load_2d(LDSOFF(lds_q), Q + qrow0 * EDIM + ec, 128, 64, EDIM);
        tdm_load_2d(LDSOFF(lds_k), K + ((size_t)b * TSEQ + j) * EDIM + ec,
                    128, 32, EDIM);
        __builtin_amdgcn_s_wait_tensorcnt(0);
      }
#else
      {
        const unsigned int* Qu = (const unsigned int*)Q;
        const unsigned int* Ku = (const unsigned int*)K;
#pragma unroll
        for (int i = 0; i < 16; ++i) {
          int li = t + 256 * i;
          int row = li >> 6, cu = li & 63;
          lds_q[row * 65 + cu] =
              Qu[(qrow0 + row) * (EDIM / 2) + ec / 2 + cu];
        }
#pragma unroll
        for (int i = 0; i < 8; ++i) {
          int li = t + 256 * i;
          int key = li >> 6, cu = li & 63;
          lds_k[key * 65 + cu] =
              Ku[((size_t)b * TSEQ + j + key) * (EDIM / 2) + ec / 2 + cu];
        }
      }
#endif
      __syncthreads();
#pragma unroll
      for (int kk = 0; kk < 4; ++kk) {
        v16bf a  = frag_a(lds_q, sm * 16, 65, kk * 16, lane);
        v16bf bb = frag_b(lds_k, sn * 16, 65, kk * 16, lane);
        s = wmma_bf16(a, bb, s);
      }
      __syncthreads();
    }
    // softmax weights -> bf16 P in LDS (re-layout C-frag -> A-frag via LDS)
    const int nl = lane & 15, half = lane >> 4;
    unsigned short* pds = (unsigned short*)lds_p;
#pragma unroll
    for (int r = 0; r < 8; ++r) {
      int row = sm * 16 + r + half * 8;
      int key = j + sn * 16 + nl;
      float sv = s[r] * 0.03125f;
      if (key > r0 + row) sv = -3.0e38f;
      float p = __expf(sv - lds_m[row]) * lds_li[row];
      pds[row * 34 + sn * 16 + nl] = f2bf(p);
    }
    __syncthreads();
    // O(64x128) += P(64x32) @ V(32x128); this wave: rows sm*16, cols sn*64..+64
    v16bf ap = frag_a(lds_p, sm * 16, 17, 0, lane);
#pragma unroll
    for (int tt = 0; tt < 4; ++tt) {
      v16bf bv = frag_b_strided((const unsigned short*)lds_v,
                                sn * 64 + tt * 16, 130, lane);
      acc[tt] = wmma_bf16(ap, bv, acc[tt]);
    }
    __syncthreads();
  }

  const int nl = lane & 15, half = lane >> 4;
#pragma unroll
  for (int tt = 0; tt < 4; ++tt)
#pragma unroll
    for (int r = 0; r < 8; ++r) {
      int row = r0 + sm * 16 + r + half * 8;
      int col = c0 + sn * 64 + tt * 16 + nl;
      out[((size_t)b * TSEQ + row) * EDIM + col] = acc[tt][r];
    }
}

// ---------------------------------------------------------------------------
extern "C" void kernel_launch(void* const* d_in, const int* in_sizes, int n_in,
                              void* d_out, int out_size, void* d_ws, size_t ws_size,
                              hipStream_t stream) {
  (void)in_sizes; (void)n_in; (void)out_size; (void)ws_size;
  const float* x  = (const float*)d_in[0];
  const float* Wq = (const float*)d_in[1];
  const float* Wk = (const float*)d_in[2];
  const float* Wv = (const float*)d_in[3];
  float* out = (float*)d_out;

  const size_t NE = (size_t)NBATCH * TSEQ * EDIM;   // 16M elems per tensor
  unsigned short* Q = (unsigned short*)d_ws;
  unsigned short* K = Q + NE;
  unsigned short* V = K + NE;
  float* Mrow = (float*)(V + NE);
  float* Lrow = Mrow + (size_t)NBATCH * TSEQ;

  // 1) Q/K/V projections (bf16 outputs into workspace)
  qkv_proj_kernel<<<dim3(128, 16, 3), 256, 0, stream>>>(x, Wq, Wk, Wv, Q, K, V);
  // 2) causal softmax statistics
  attn_stats_kernel<<<dim3(32, 8), 256, 0, stream>>>(Q, K, Mrow, Lrow);
  // 3) attention output
  attn_out_kernel<<<dim3(8, 32, 8), 256, 0, stream>>>(Q, K, V, Mrow, Lrow, out);
}
